// RWKV_TimeMix_50921132261551
// MI455X (gfx1250) — matrix operand order repaired and checked by reading, not compiled, not run on hardware
//
#include <hip/hip_runtime.h>
#include <hip/hip_bf16.h>

// ---------------------------------------------------------------------------
// RWKV TimeMix for MI455X (gfx1250): bf16 WMMA GEMMs + fused WKV recurrence.
// B=8, T=1024, C=2048.
// Round 2: per-wave GEMM tile 32x32 -> 64x32 (8 WMMA accumulators),
//          block tile 128x128; 21.3 flop per L0 byte (was 16).
// ---------------------------------------------------------------------------

typedef __bf16 bf16_t;
typedef __attribute__((ext_vector_type(16))) __bf16 v16bf;
typedef __attribute__((ext_vector_type(8)))  float  v8f;

static constexpr int RB = 8;
static constexpr int RT = 1024;
static constexpr int RC = 2048;
static constexpr int RM = RB * RT;          // 8192 rows (b,t flattened)

// ---------------------------------------------------------------------------
// 1) f32 -> bf16 conversion (weights)
// ---------------------------------------------------------------------------
__global__ void cvt_f32_to_bf16_kernel(const float* __restrict__ in,
                                       bf16_t* __restrict__ out, int n) {
  int i = blockIdx.x * blockDim.x + threadIdx.x;
  if (i < n) out[i] = (bf16_t)in[i];
}

// ---------------------------------------------------------------------------
// 2) time-shift mixing: xk/xv/xr = x*mix + shift(x)*(1-mix), stored bf16
// ---------------------------------------------------------------------------
__global__ void mix_kernel(const float* __restrict__ x,
                           const float* __restrict__ tmk,
                           const float* __restrict__ tmv,
                           const float* __restrict__ tmr,
                           bf16_t* __restrict__ xk,
                           bf16_t* __restrict__ xv,
                           bf16_t* __restrict__ xr) {
  size_t idx = (size_t)blockIdx.x * blockDim.x + threadIdx.x;
  size_t total = (size_t)RB * RT * RC;
  if (idx >= total) return;
  int c = (int)(idx % RC);
  int t = (int)((idx / RC) % RT);
  float xc = x[idx];
  float xp = (t > 0) ? x[idx - RC] : 0.0f;
  float mk = tmk[c], mv = tmv[c], mr = tmr[c];
  xk[idx] = (bf16_t)(xc * mk + xp * (1.0f - mk));
  xv[idx] = (bf16_t)(xc * mv + xp * (1.0f - mv));
  xr[idx] = (bf16_t)(xc * mr + xp * (1.0f - mr));
}

// ---------------------------------------------------------------------------
// 3) bf16 WMMA GEMM:  O[m,n] = sum_c A[m,c] * W[n,c]
//    A: [M,K] row-major bf16, W: [N,K] row-major bf16, O: [M,N] f32.
//    Contraction dim c is contiguous for BOTH operands -> no transpose needed:
//      A-frag (16x32): lane = row m0+lane%16; K-halves {0-7,16-23}/{8-15,24-31}
//      B-frag (32x16): lane = W row n0+lane%16; K-halves {0-7,16-23}/{8-15,24-31}
//    Block tile 128x128: 8 waves (2 along M x 4 along N), 64x32 per wave
//    = 4x2 WMMA tiles, 8 f32 accumulators (64 VGPRs).
// ---------------------------------------------------------------------------
__global__ __launch_bounds__(256) void gemm_bf16_wmma_kernel(
    const bf16_t* __restrict__ A0, const bf16_t* __restrict__ W0, float* __restrict__ O0,
    const bf16_t* __restrict__ A1, const bf16_t* __restrict__ W1, float* __restrict__ O1,
    const bf16_t* __restrict__ A2, const bf16_t* __restrict__ W2, float* __restrict__ O2,
    int M, int N, int K) {
  const bf16_t* A; const bf16_t* W; float* O;
  switch (blockIdx.z) {
    case 0:  A = A0; W = W0; O = O0; break;
    case 1:  A = A1; W = W1; O = O1; break;
    default: A = A2; W = W2; O = O2; break;
  }

  const int lane   = threadIdx.x & 31;   // wave32
  const int wid    = threadIdx.x >> 5;   // 0..7
  const int wm     = wid & 1;            // 2 waves along M
  const int wn     = wid >> 1;           // 4 waves along N
  const int m0     = blockIdx.y * 128 + wm * 64;
  const int n0     = blockIdx.x * 128 + wn * 32;
  const int half   = lane >> 4;          // 0: lanes 0-15, 1: lanes 16-31
  const int laneM  = lane & 15;
  const int aSel   = half * 8;           // A K-offset within 32-chunk
  const int bSel   = half * 8;           // B K-offset within 32-chunk

  union Frag { v16bf v; uint4 q[2]; };

  v8f acc[4][2];
#pragma unroll
  for (int i = 0; i < 4; ++i)
#pragma unroll
    for (int j = 0; j < 2; ++j) acc[i][j] = {};

  const bf16_t* aRow[4];
  const bf16_t* bRow[2];
#pragma unroll
  for (int i = 0; i < 4; ++i) aRow[i] = A + (size_t)(m0 + i * 16 + laneM) * K;
#pragma unroll
  for (int j = 0; j < 2; ++j) bRow[j] = W + (size_t)(n0 + j * 16 + laneM) * K;

#pragma unroll 2
  for (int c0 = 0; c0 < K; c0 += 32) {
    Frag a[4], b[2];
    // A fragments: two 16B contiguous loads per lane per tile
#pragma unroll
    for (int i = 0; i < 4; ++i) {
      a[i].q[0] = *(const uint4*)(aRow[i] + c0 + aSel);
      a[i].q[1] = *(const uint4*)(aRow[i] + c0 + 16 + aSel);
    }
    // B fragments: two 16B contiguous loads per lane per tile
#pragma unroll
    for (int j = 0; j < 2; ++j) {
      b[j].q[0] = *(const uint4*)(bRow[j] + c0 + bSel);
      b[j].q[1] = *(const uint4*)(bRow[j] + c0 + 16 + bSel);
    }
#pragma unroll
    for (int i = 0; i < 4; ++i)
#pragma unroll
      for (int j = 0; j < 2; ++j)
        acc[i][j] = __builtin_amdgcn_wmma_f32_16x16x32_bf16(
            false, a[i].v, false, b[j].v, (short)0, acc[i][j], false, false);
  }

  // C/D layout: VGPR r -> row (r + 8*half), lane%16 -> column
  const int col  = laneM;
  const int rOff = half * 8;
#pragma unroll
  for (int i = 0; i < 4; ++i)
#pragma unroll
    for (int j = 0; j < 2; ++j)
#pragma unroll
      for (int r = 0; r < 8; ++r)
        O[(size_t)(m0 + i * 16 + rOff + r) * N + (n0 + j * 16 + col)] = acc[i][j][r];
}

// ---------------------------------------------------------------------------
// 4) WKV recurrence (numerically stable, matches reference) fused with
//    sigmoid(r) gating; output y = sr*wkv in bf16. One thread per (b,c)
//    channel, sequential over T, loads coalesced across c.
// ---------------------------------------------------------------------------
__global__ void wkv_kernel(const float* __restrict__ k,
                           const float* __restrict__ v,
                           const float* __restrict__ r,
                           const float* __restrict__ time_decay,
                           const float* __restrict__ time_first,
                           bf16_t* __restrict__ y) {
  int idx = blockIdx.x * blockDim.x + threadIdx.x;
  if (idx >= RB * RC) return;
  const int b = idx / RC;
  const int c = idx % RC;

  const float w = -__expf(time_decay[c]);
  const float u = time_first[c];

  float aa = 0.0f, bb = 0.0f, pp = -1e38f;

  size_t base = (size_t)b * RT * RC + c;
  for (int t = 0; t < RT; ++t, base += RC) {
    const float kt = k[base];
    const float vt = v[base];
    const float rt = r[base];

    float ww = u + kt;
    float p  = fmaxf(pp, ww);
    float e1 = __expf(pp - p);
    float e2 = __expf(ww - p);
    float num = e1 * aa + e2 * vt;
    float den = fmaxf(e1 * bb + e2, 1e-9f);
    float yv  = num / den;

    float sr = 1.0f / (1.0f + __expf(-rt));
    y[base] = (bf16_t)(sr * yv);

    float ww2 = w + pp;
    float p2  = fmaxf(ww2, kt);
    float e1b = __expf(ww2 - p2);
    float e2b = __expf(kt - p2);
    aa = e1b * aa + e2b * vt;
    bb = e1b * bb + e2b;
    pp = p2;
  }
}

// ---------------------------------------------------------------------------
// Host-side launcher
// ---------------------------------------------------------------------------
extern "C" void kernel_launch(void* const* d_in, const int* in_sizes, int n_in,
                              void* d_out, int out_size, void* d_ws, size_t ws_size,
                              hipStream_t stream) {
  const float* x    = (const float*)d_in[0];
  const float* td   = (const float*)d_in[1];
  const float* tf   = (const float*)d_in[2];
  const float* tmk  = (const float*)d_in[3];
  const float* tmv  = (const float*)d_in[4];
  const float* tmr  = (const float*)d_in[5];
  const float* Wk   = (const float*)d_in[6];
  const float* Wv   = (const float*)d_in[7];
  const float* Wr   = (const float*)d_in[8];
  const float* Wo   = (const float*)d_in[9];
  float* out        = (float*)d_out;

  const size_t S_btc_bf = (size_t)RM * RC * sizeof(bf16_t);   // 32 MiB
  const size_t S_cc_bf  = (size_t)RC * RC * sizeof(bf16_t);   //  8 MiB
  const size_t S_btc_f  = (size_t)RM * RC * sizeof(float);    // 64 MiB

  char* p = (char*)d_ws;
  bf16_t* xk_b = (bf16_t*)(p);
  bf16_t* xv_b = (bf16_t*)(p + S_btc_bf);
  bf16_t* xr_b = (bf16_t*)(p + 2 * S_btc_bf);
  bf16_t* Wk_b = (bf16_t*)(p + 3 * S_btc_bf);
  bf16_t* Wv_b = (bf16_t*)(p + 3 * S_btc_bf + 1 * S_cc_bf);
  bf16_t* Wr_b = (bf16_t*)(p + 3 * S_btc_bf + 2 * S_cc_bf);
  bf16_t* Wo_b = (bf16_t*)(p + 3 * S_btc_bf + 3 * S_cc_bf);
  float*  kf   = (float*)(p + 3 * S_btc_bf + 4 * S_cc_bf);
  float*  vf   = (float*)(p + 3 * S_btc_bf + 4 * S_cc_bf + 1 * S_btc_f);
  float*  rf   = (float*)(p + 3 * S_btc_bf + 4 * S_cc_bf + 2 * S_btc_f);
  bf16_t* y_b  = xk_b;  // xk no longer needed once k-GEMM is done

  // 1) weights -> bf16
  {
    int n = RC * RC;
    int blocks = (n + 255) / 256;
    cvt_f32_to_bf16_kernel<<<blocks, 256, 0, stream>>>(Wk, Wk_b, n);
    cvt_f32_to_bf16_kernel<<<blocks, 256, 0, stream>>>(Wv, Wv_b, n);
    cvt_f32_to_bf16_kernel<<<blocks, 256, 0, stream>>>(Wr, Wr_b, n);
    cvt_f32_to_bf16_kernel<<<blocks, 256, 0, stream>>>(Wo, Wo_b, n);
  }

  // 2) time-shift mix
  {
    size_t total = (size_t)RM * RC;
    int blocks = (int)((total + 255) / 256);
    mix_kernel<<<blocks, 256, 0, stream>>>(x, tmk, tmv, tmr, xk_b, xv_b, xr_b);
  }

  // 3) k = xk@Wk^T, v = xv@Wv^T, r = xr@Wr^T (one kernel, z selects)
  {
    dim3 grid(RC / 128, RM / 128, 3);
    gemm_bf16_wmma_kernel<<<grid, 256, 0, stream>>>(
        xk_b, Wk_b, kf,
        xv_b, Wv_b, vf,
        xr_b, Wr_b, rf,
        RM, RC, RC);
  }

  // 4) WKV recurrence + sigmoid gate -> y (bf16)
  {
    int n = RB * RC;
    wkv_kernel<<<(n + 255) / 256, 256, 0, stream>>>(kf, vf, rf, td, tf, y_b);
  }

  // 5) out = y @ Wo^T (f32 result straight to d_out)
  {
    dim3 grid(RC / 128, RM / 128, 1);
    gemm_bf16_wmma_kernel<<<grid, 256, 0, stream>>>(
        y_b, Wo_b, out,
        y_b, Wo_b, out,
        y_b, Wo_b, out,
        RM, RC, RC);
  }
}
